// MoeLayer_47639777247356
// MI455X (gfx1250) — compile-verified, matrix operand order checked by reading, not run on hardware
//
#include <hip/hip_runtime.h>
#include <hip/hip_bf16.h>

// ---------------- problem constants ----------------
static constexpr int BB = 4, TT = 2048, CC = 1024, HH = 2048, EE = 8;
static constexpr int NTOK = BB * TT;                 // 8192 tokens
static constexpr int MTILE = 64;                     // rows per block tile
static constexpr int RCAP = 2 * NTOK + EE * MTILE;   // padded expanded-row capacity = 16896
static constexpr int MT   = RCAP / MTILE;            // 264 row tiles

typedef __bf16 bf16;
typedef __attribute__((ext_vector_type(16))) __bf16 v16bf;
typedef __attribute__((ext_vector_type(8)))  __bf16 v8bf;
typedef __attribute__((ext_vector_type(8)))  float  v8f;

union FragBF { v16bf v; v8bf h[2]; };

// ---------------- workspace layout (bytes) ----------------
static constexpr size_t O_XB  = 0;                                     // x bf16   [NTOK*CC]
static constexpr size_t O_WG  = O_XB  + (size_t)NTOK * CC * 2;         // gate_proj bf16 [E*H*C]
static constexpr size_t O_WU  = O_WG  + (size_t)EE * HH * CC * 2;
static constexpr size_t O_WD  = O_WU  + (size_t)EE * HH * CC * 2;      // down_proj bf16 [E*C*H]
static constexpr size_t O_H   = O_WD  + (size_t)EE * CC * HH * 2;      // hidden bf16 [RCAP*H]
static constexpr size_t O_TOK = O_H   + (size_t)RCAP * HH * 2;         // int  [RCAP]
static constexpr size_t O_WT  = O_TOK + (size_t)RCAP * 4;              // float[RCAP]
static constexpr size_t O_CNT = O_WT  + (size_t)RCAP * 4;              // int  [8]
static constexpr size_t O_OFF = O_CNT + 8 * 4;                         // int  [9]
static constexpr size_t O_CUR = O_OFF + 9 * 4;                         // int  [8]
static constexpr size_t O_EIX = ((O_CUR + 8 * 4 + 255) / 256) * 256;   // int  [NTOK*2]
static constexpr size_t O_EP  = O_EIX + (size_t)NTOK * 2 * 4;          // float[NTOK*2]

// ---------------- CDNA5 async-copy helpers ----------------
__device__ __forceinline__ void async_fill_b128(void* ldsDst, const void* gsrc) {
    unsigned lds = (unsigned)(unsigned long long)ldsDst;   // low 32 bits = LDS offset
    asm volatile("global_load_async_to_lds_b128 %0, %1, off"
                 :: "v"(lds), "v"(gsrc) : "memory");
}
__device__ __forceinline__ void wait_async0() {
    asm volatile("s_wait_asynccnt 0x0" ::: "memory");
}

// ---------------- utility kernels ----------------
__global__ void k_zero_f(float* p, int n) {
    int i = blockIdx.x * blockDim.x + threadIdx.x;
    for (; i < n; i += gridDim.x * blockDim.x) p[i] = 0.0f;
}
__global__ void k_zero_i(int* p, int n) {
    int i = blockIdx.x * blockDim.x + threadIdx.x;
    for (; i < n; i += gridDim.x * blockDim.x) p[i] = 0;
}
__global__ void k_cast_bf16(const float* __restrict__ src, bf16* __restrict__ dst, int n) {
    int i = blockIdx.x * blockDim.x + threadIdx.x;
    for (; i < n; i += gridDim.x * blockDim.x) dst[i] = (bf16)src[i];
}

// ---------------- gating: logits -> top2 -> softmax -> counts ----------------
__global__ void k_gate(const float* __restrict__ x, const float* __restrict__ wg,
                       int* __restrict__ eidx, float* __restrict__ ep, int* __restrict__ cnt) {
    const int lane = threadIdx.x & 31;
    const int wave = threadIdx.x >> 5;
    const int t = blockIdx.x * 8 + wave;
    if (t >= NTOK) return;

    float acc[EE];
#pragma unroll
    for (int e = 0; e < EE; ++e) acc[e] = 0.0f;

    const float* xr = x + (size_t)t * CC;
    for (int c = lane; c < CC; c += 32) {
        float xv = xr[c];
#pragma unroll
        for (int e = 0; e < EE; ++e) acc[e] += xv * wg[e * CC + c];
    }
#pragma unroll
    for (int e = 0; e < EE; ++e) {
#pragma unroll
        for (int off = 16; off > 0; off >>= 1)
            acc[e] += __shfl_xor(acc[e], off, 32);
    }
    if (lane == 0) {
        float best = -1e30f; int bi = 0;
#pragma unroll
        for (int e = 0; e < EE; ++e) if (acc[e] > best) { best = acc[e]; bi = e; }
        float sec = -1e30f; int si = (bi == 0) ? 1 : 0;
#pragma unroll
        for (int e = 0; e < EE; ++e) if (e != bi && acc[e] > sec) { sec = acc[e]; si = e; }
        float e1 = __expf(sec - best);
        float inv = 1.0f / (1.0f + e1);
        eidx[2 * t + 0] = bi; ep[2 * t + 0] = inv;
        eidx[2 * t + 1] = si; ep[2 * t + 1] = e1 * inv;
        atomicAdd(&cnt[bi], 1);
        atomicAdd(&cnt[si], 1);
    }
}

// ---------------- per-expert offsets (padded to MTILE) ----------------
__global__ void k_offsets(const int* __restrict__ cnt, int* __restrict__ off, int* __restrict__ cur) {
    if (threadIdx.x == 0) {
        int run = 0;
        for (int e = 0; e < EE; ++e) {
            off[e] = run;
            run += ((cnt[e] + MTILE - 1) / MTILE) * MTILE;
        }
        off[EE] = run;
        for (int e = 0; e < EE; ++e) cur[e] = 0;
    }
}

// ---------------- scatter token -> expanded rows ----------------
__global__ void k_scatter(const int* __restrict__ eidx, const float* __restrict__ ep,
                          const int* __restrict__ off, int* __restrict__ cur,
                          int* __restrict__ tok, float* __restrict__ wt) {
    int t = blockIdx.x * blockDim.x + threadIdx.x;
    if (t >= NTOK) return;
#pragma unroll
    for (int k = 0; k < 2; ++k) {
        int e = eidx[2 * t + k];
        float p = ep[2 * t + k];
        int pos = atomicAdd(&cur[e], 1);
        int row = off[e] + pos;
        tok[row] = t;
        wt[row] = p;
    }
}

// ---------------- WMMA helpers ----------------
__device__ __forceinline__ v8f wmma_bf16(const v16bf& a, const v16bf& b, v8f c) {
    return __builtin_amdgcn_wmma_f32_16x16x32_bf16(false, a, false, b, (short)0, c, false, false);
}

__device__ __forceinline__ FragBF load_frag_lds(const bf16* p) {
    FragBF f;
    f.h[0] = *(const v8bf*)p;
    f.h[1] = *(const v8bf*)(p + 16);
    return f;
}

// one K=32 step of the dual (gate/up) GEMM, 4 M-subtiles
__device__ __forceinline__ void ffn1_step(const bf16 (*buf)[32], int m, int kb,
                                          const bf16* g0, const bf16* u0,
                                          v8f& g0a, v8f& g1a, v8f& g2a, v8f& g3a,
                                          v8f& u0a, v8f& u1a, v8f& u2a, v8f& u3a) {
    FragBF bg = load_frag_lds(g0);          // actually global, same codegen path
    FragBF bu = load_frag_lds(u0);
    FragBF a;
    a = load_frag_lds(&buf[0 * 16 + m][kb]);
    g0a = wmma_bf16(a.v, bg.v, g0a); u0a = wmma_bf16(a.v, bu.v, u0a);
    a = load_frag_lds(&buf[1 * 16 + m][kb]);
    g1a = wmma_bf16(a.v, bg.v, g1a); u1a = wmma_bf16(a.v, bu.v, u1a);
    a = load_frag_lds(&buf[2 * 16 + m][kb]);
    g2a = wmma_bf16(a.v, bg.v, g2a); u2a = wmma_bf16(a.v, bu.v, u2a);
    a = load_frag_lds(&buf[3 * 16 + m][kb]);
    g3a = wmma_bf16(a.v, bg.v, g3a); u3a = wmma_bf16(a.v, bu.v, u3a);
}

// one K=32 step of a single GEMM, 4 M-subtiles
__device__ __forceinline__ void ffn2_step(const bf16 (*buf)[32], int m, int kb,
                                          const bf16* d0,
                                          v8f& a0, v8f& a1, v8f& a2, v8f& a3) {
    FragBF b = load_frag_lds(d0);
    FragBF a;
    a = load_frag_lds(&buf[0 * 16 + m][kb]); a0 = wmma_bf16(a.v, b.v, a0);
    a = load_frag_lds(&buf[1 * 16 + m][kb]); a1 = wmma_bf16(a.v, b.v, a1);
    a = load_frag_lds(&buf[2 * 16 + m][kb]); a2 = wmma_bf16(a.v, b.v, a2);
    a = load_frag_lds(&buf[3 * 16 + m][kb]); a3 = wmma_bf16(a.v, b.v, a3);
}

// ---------------- FFN stage 1: h = silu(x Wg^T) * (x Wu^T) ----------------
// grid: (MT, H/128), block 256 (8 waves). Block tile: M=64, N=128.
__global__ void __launch_bounds__(256, 1)
k_ffn1(const bf16* __restrict__ xb, const bf16* __restrict__ wgb,
       const bf16* __restrict__ wub, const int* __restrict__ tok,
       const int* __restrict__ off, bf16* __restrict__ hb) {
    __shared__ __align__(16) bf16 lA[2][MTILE][32];    // double-buffered A tile (8 KB)

    const int rowbase = blockIdx.x * MTILE;
    const int total = off[EE];
    if (rowbase >= total) return;

    int e = 0;
#pragma unroll
    for (int i = 0; i < EE; ++i) if (rowbase >= off[i + 1]) e = i + 1;

    // per-thread A-fill source (loop-invariant gather base, advanced by K-step)
    const int fr = threadIdx.x >> 2;                   // row 0..63
    const int fs = threadIdx.x & 3;                    // 16B segment 0..3
    const int tokRow = tok[rowbase + fr];
    const bf16* asrc = xb + (size_t)tokRow * CC + fs * 8;
    bf16* ldst0 = &lA[0][fr][fs * 8];
    bf16* ldst1 = &lA[1][fr][fs * 8];

    const int lane = threadIdx.x & 31;
    const int wave = threadIdx.x >> 5;
    const int col = blockIdx.y * 128 + wave * 16 + (lane & 15);   // h index
    const int kb = (lane >> 4) * 8;                               // K band select
    const int m = lane & 15;

    const bf16* bgp = wgb + ((size_t)e * HH + col) * CC + kb;
    const bf16* bup = wub + ((size_t)e * HH + col) * CC + kb;

    v8f aG0 = {}, aG1 = {}, aG2 = {}, aG3 = {};
    v8f aU0 = {}, aU1 = {}, aU2 = {}, aU3 = {};

    // prologue: async-fill buffer 0
    async_fill_b128(ldst0, asrc);
    wait_async0();
    __syncthreads();

    const int nsteps = CC / 32;                        // 32, even
    for (int s = 0; s < nsteps; s += 2) {
        // --- even step: compute buf0, prefetch buf1 ---
        async_fill_b128(ldst1, asrc + (s + 1) * 32);
        ffn1_step(lA[0], m, kb, bgp + s * 32, bup + s * 32,
                  aG0, aG1, aG2, aG3, aU0, aU1, aU2, aU3);
        wait_async0();
        __syncthreads();
        // --- odd step: compute buf1, prefetch buf0 ---
        if (s + 2 < nsteps) async_fill_b128(ldst0, asrc + (s + 2) * 32);
        ffn1_step(lA[1], m, kb, bgp + (s + 1) * 32, bup + (s + 1) * 32,
                  aG0, aG1, aG2, aG3, aU0, aU1, aU2, aU3);
        wait_async0();
        __syncthreads();
    }

    const int mrow = (lane >> 4) * 8;
#define FFN1_EPI(AG, AU, S4)                                            \
    _Pragma("unroll")                                                   \
    for (int i = 0; i < 8; ++i) {                                       \
        float g = AG[i], u = AU[i];                                     \
        float hv = g * (1.0f / (1.0f + __expf(-g))) * u;                \
        hb[(size_t)(rowbase + (S4) * 16 + mrow + i) * HH + col] = (bf16)hv; \
    }
    FFN1_EPI(aG0, aU0, 0)
    FFN1_EPI(aG1, aU1, 1)
    FFN1_EPI(aG2, aU2, 2)
    FFN1_EPI(aG3, aU3, 3)
#undef FFN1_EPI
}

// ---------------- FFN stage 2: out += w * (h Wd^T), scatter-add ----------------
// grid: (MT, C/128), block 256 (8 waves). Block tile: M=64, N=128.
__global__ void __launch_bounds__(256, 1)
k_ffn2(const bf16* __restrict__ hb, const bf16* __restrict__ wdb,
       const int* __restrict__ tok, const float* __restrict__ wt,
       const int* __restrict__ off, float* __restrict__ out) {
    __shared__ __align__(16) bf16 lA[2][MTILE][32];
    __shared__ int   sTok[MTILE];
    __shared__ float sWt[MTILE];

    const int rowbase = blockIdx.x * MTILE;
    const int total = off[EE];
    if (rowbase >= total) return;

    int e = 0;
#pragma unroll
    for (int i = 0; i < EE; ++i) if (rowbase >= off[i + 1]) e = i + 1;

    if (threadIdx.x < MTILE) {
        sTok[threadIdx.x] = tok[rowbase + threadIdx.x];
        sWt[threadIdx.x] = wt[rowbase + threadIdx.x];
    }

    const int fr = threadIdx.x >> 2;
    const int fs = threadIdx.x & 3;
    const bf16* asrc = hb + (size_t)(rowbase + fr) * HH + fs * 8;
    bf16* ldst0 = &lA[0][fr][fs * 8];
    bf16* ldst1 = &lA[1][fr][fs * 8];

    const int lane = threadIdx.x & 31;
    const int wave = threadIdx.x >> 5;
    const int col = blockIdx.y * 128 + wave * 16 + (lane & 15);   // c index
    const int kb = (lane >> 4) * 8;
    const int m = lane & 15;

    const bf16* bdp = wdb + ((size_t)e * CC + col) * HH + kb;

    v8f a0 = {}, a1 = {}, a2 = {}, a3 = {};

    async_fill_b128(ldst0, asrc);
    wait_async0();
    __syncthreads();

    const int nsteps = HH / 32;                        // 64, even
    for (int s = 0; s < nsteps; s += 2) {
        async_fill_b128(ldst1, asrc + (s + 1) * 32);
        ffn2_step(lA[0], m, kb, bdp + s * 32, a0, a1, a2, a3);
        wait_async0();
        __syncthreads();
        if (s + 2 < nsteps) async_fill_b128(ldst0, asrc + (s + 2) * 32);
        ffn2_step(lA[1], m, kb, bdp + (s + 1) * 32, a0, a1, a2, a3);
        wait_async0();
        __syncthreads();
    }

    const int mrow = (lane >> 4) * 8;
#define FFN2_EPI(ACC, S4)                                               \
    _Pragma("unroll")                                                   \
    for (int i = 0; i < 8; ++i) {                                       \
        int mi = (S4) * 16 + mrow + i;                                  \
        float val = ACC[i] * sWt[mi];                                   \
        atomicAdd(&out[(size_t)sTok[mi] * CC + col], val);              \
    }
    FFN2_EPI(a0, 0)
    FFN2_EPI(a1, 1)
    FFN2_EPI(a2, 2)
    FFN2_EPI(a3, 3)
#undef FFN2_EPI
}

// ---------------- launcher ----------------
extern "C" void kernel_launch(void* const* d_in, const int* in_sizes, int n_in,
                              void* d_out, int out_size, void* d_ws, size_t ws_size,
                              hipStream_t stream) {
    const float* x   = (const float*)d_in[0];
    const float* wg  = (const float*)d_in[1];   // [E,C]
    const float* wgp = (const float*)d_in[2];   // [E,H,C]
    const float* wup = (const float*)d_in[3];   // [E,H,C]
    const float* wdp = (const float*)d_in[4];   // [E,C,H]
    float* out = (float*)d_out;

    char* ws = (char*)d_ws;
    bf16*  xb   = (bf16*)(ws + O_XB);
    bf16*  wgb  = (bf16*)(ws + O_WG);
    bf16*  wub  = (bf16*)(ws + O_WU);
    bf16*  wdb  = (bf16*)(ws + O_WD);
    bf16*  hb   = (bf16*)(ws + O_H);
    int*   tok  = (int*)(ws + O_TOK);
    float* wt   = (float*)(ws + O_WT);
    int*   cnt  = (int*)(ws + O_CNT);
    int*   off  = (int*)(ws + O_OFF);
    int*   cur  = (int*)(ws + O_CUR);
    int*   eidx = (int*)(ws + O_EIX);
    float* ep   = (float*)(ws + O_EP);

    // zero output + routing region (tok, wt, cnt contiguous ints)
    k_zero_f<<<2048, 256, 0, stream>>>(out, NTOK * CC);
    k_zero_i<<<(RCAP * 2 + 8 + 255) / 256, 256, 0, stream>>>(tok, RCAP * 2 + 8);

    // casts to bf16 (weights become L2-resident at ~100MB total)
    k_cast_bf16<<<2048, 256, 0, stream>>>(x,   xb,  NTOK * CC);
    k_cast_bf16<<<4096, 256, 0, stream>>>(wgp, wgb, EE * HH * CC);
    k_cast_bf16<<<4096, 256, 0, stream>>>(wup, wub, EE * HH * CC);
    k_cast_bf16<<<4096, 256, 0, stream>>>(wdp, wdb, EE * CC * HH);

    // routing
    k_gate<<<NTOK / 8, 256, 0, stream>>>(x, wg, eidx, ep, cnt);
    k_offsets<<<1, 32, 0, stream>>>(cnt, off, cur);
    k_scatter<<<NTOK / 256, 256, 0, stream>>>(eidx, ep, off, cur, tok, wt);

    // grouped-GEMM FFN
    k_ffn1<<<dim3(MT, HH / 128), 256, 0, stream>>>(xb, wgb, wub, tok, off, hb);
    k_ffn2<<<dim3(MT, CC / 128), 256, 0, stream>>>(hb, wdb, tok, wt, off, out);
}
